// BEVDeformableTransformerEncoder_86036784873799
// MI455X (gfx1250) — matrix operand order repaired and checked by reading, not compiled, not run on hardware
//
#include <hip/hip_runtime.h>
#include <hip/hip_bf16.h>
#include <math.h>

typedef __attribute__((ext_vector_type(16))) _Float16 v16h;
typedef __attribute__((ext_vector_type(8)))  float    v8f;

#define HW    1600   // 40*40 pixels
#define NH    8      // heads == groups
#define DH    64     // dim_head
#define CG    32     // channels per group (256/8)
#define NJ    100    // 10*10 sampled keys
#define JP    112    // j padded to 7*16 for sim N-tiles
#define JK    128    // j padded to 4*32 for attn@v K-steps
#define SCALE 0.125f // 64^-0.5
#define CPB_TPW 8    // CPB tiles (16 points each) per wave

__device__ __forceinline__ v8f wmma16(v16h a, v16h b, v8f c) {
  return __builtin_amdgcn_wmma_f32_16x16x32_f16(false, a, false, b, (short)0, c, false, false);
}
// A-fragment K index for 16-bit A 16x32 (contiguous, matches B fills): half*16+2v
__device__ __forceinline__ int akidx(int v, int half) {
  return half * 16 + 2 * v;
}

// ---------------- K1: grouped 1x1 q projection; writes q_f32 [oc][p] and q*SCALE f16 [h][i][d]
__global__ void k_proj_q(const float* __restrict__ x, const float* __restrict__ wq,
                         float* __restrict__ qf, _Float16* __restrict__ qs16) {
  int idx = blockIdx.x * 256 + threadIdx.x;          // 512*1600
  int oc = idx / HW, p = idx % HW;
  int g = oc >> 6, d = oc & 63;
  float acc = 0.f;
  for (int ic = 0; ic < CG; ++ic)
    acc += wq[oc * CG + ic] * x[(g * CG + ic) * HW + p];
  qf[idx] = acc;
  qs16[(g * HW + p) * DH + d] = (_Float16)(acc * SCALE);
}

// ---------------- K2: offset net (depthwise 6x6 s4 + GELU -> 1x1 -> tanh*4) -> normalized vgrid
__global__ void k_offsets(const float* __restrict__ q, const float* __restrict__ w1dw,
                          const float* __restrict__ b1dw, const float* __restrict__ w2off,
                          float* __restrict__ vgrid) {
  __shared__ float sh[64];
  __shared__ float off2[2];
  int blk = blockIdx.x;                 // 8 groups * 100 output points
  int g = blk / 100, oh = (blk % 100) / 10, ow = blk % 10;
  int ch = threadIdx.x;                 // 64 channels
  float acc = 0.f;
  for (int ky = 0; ky < 6; ++ky) {
    int ih = oh * 4 - 1 + ky;
    if (ih < 0 || ih >= 40) continue;
    for (int kx = 0; kx < 6; ++kx) {
      int iw = ow * 4 - 1 + kx;
      if (iw < 0 || iw >= 40) continue;
      acc += w1dw[ch * 36 + ky * 6 + kx] * q[(g * 64 + ch) * HW + ih * 40 + iw];
    }
  }
  float vv = acc + b1dw[ch];
  sh[ch] = 0.5f * vv * (1.0f + erff(vv * 0.70710678118654752f));  // exact GELU
  __syncthreads();
  if (ch < 2) {
    float o = 0.f;
    for (int c2 = 0; c2 < 64; ++c2) o += w2off[ch * 64 + c2] * sh[c2];
    off2[ch] = tanhf(o) * 4.0f;
  }
  __syncthreads();
  if (ch == 0) {
    int j = oh * 10 + ow;
    float vx = (float)ow + off2[0];
    float vy = (float)oh + off2[1];
    vgrid[(g * NJ + j) * 2 + 0] = 2.0f * vx / 9.0f - 1.0f;
    vgrid[(g * NJ + j) * 2 + 1] = 2.0f * vy / 9.0f - 1.0f;
  }
}

// ---------------- K3: bilinear grid_sample (zeros padding, align_corners=False) -> kv [c][j]
__global__ void k_sample(const float* __restrict__ x, const float* __restrict__ vgrid,
                         float* __restrict__ kv) {
  int idx = blockIdx.x * 256 + threadIdx.x;          // 8*100*32
  if (idx >= NH * NJ * CG) return;
  int g = idx / (NJ * CG), j = (idx / CG) % NJ, cc = idx % CG;
  float gx = ((vgrid[(g * NJ + j) * 2 + 0] + 1.0f) * 40.0f - 1.0f) * 0.5f;
  float gy = ((vgrid[(g * NJ + j) * 2 + 1] + 1.0f) * 40.0f - 1.0f) * 0.5f;
  float x0 = floorf(gx), y0 = floorf(gy);
  float wx = gx - x0, wy = gy - y0;
  const float* img = x + (g * CG + cc) * HW;
  float v00 = 0.f, v10 = 0.f, v01 = 0.f, v11 = 0.f;
  int xi = (int)x0, yi = (int)y0;
  if (x0 >= 0.f && x0 <= 39.f && y0 >= 0.f && y0 <= 39.f) v00 = img[yi * 40 + xi];
  if (x0 + 1.f >= 0.f && x0 + 1.f <= 39.f && y0 >= 0.f && y0 <= 39.f) v10 = img[yi * 40 + xi + 1];
  if (x0 >= 0.f && x0 <= 39.f && y0 + 1.f >= 0.f && y0 + 1.f <= 39.f) v01 = img[(yi + 1) * 40 + xi];
  if (x0 + 1.f >= 0.f && x0 + 1.f <= 39.f && y0 + 1.f >= 0.f && y0 + 1.f <= 39.f) v11 = img[(yi + 1) * 40 + xi + 1];
  kv[(g * CG + cc) * NJ + j] = v00 * (1 - wx) * (1 - wy) + v10 * wx * (1 - wy)
                             + v01 * (1 - wx) * wy + v11 * wx * wy;
}

// ---------------- K4: grouped 1x1 k/v projections -> k16 [h][jp112][d], vt16 [h][d][jk128]
__global__ void k_proj_kv(const float* __restrict__ kv, const float* __restrict__ wk,
                          const float* __restrict__ wv, _Float16* __restrict__ k16,
                          _Float16* __restrict__ vt16) {
  const int NK = 512 * JP, NV = 512 * JK;
  int idx = blockIdx.x * 256 + threadIdx.x;
  if (idx < NK) {
    int oc = idx / JP, j = idx % JP;
    int g = oc >> 6, d = oc & 63;
    float acc = 0.f;
    if (j < NJ)
      for (int ic = 0; ic < CG; ++ic) acc += wk[oc * CG + ic] * kv[(g * CG + ic) * NJ + j];
    k16[(g * JP + j) * DH + d] = (_Float16)acc;
  } else if (idx < NK + NV) {
    int t = idx - NK;
    int oc = t / JK, j = t % JK;
    int g = oc >> 6, d = oc & 63;
    float acc = 0.f;
    if (j < NJ)
      for (int ic = 0; ic < CG; ++ic) acc += wv[oc * CG + ic] * kv[(g * CG + ic) * NJ + j];
    vt16[(g * DH + d) * JK + j] = (_Float16)acc;
  }
}

// ---------------- K5: CPB bias MLP via WMMA. 1.28M points; 16-point tiles, 8 tiles/wave so the
// register-resident W1^T B-fragments and b1/w2 scalars are reused across 64 WMMAs per wave.
// launch_bounds(128,1): let the allocator keep all fragments in VGPRs (no scratch spills).
__global__ void __launch_bounds__(128, 1)
k_cpb_bias(const float* __restrict__ vgrid,
           const float* __restrict__ w0, const float* __restrict__ b0,
           const float* __restrict__ w1, const float* __restrict__ b1,
           const float* __restrict__ w2, const float* __restrict__ b2,
           float* __restrict__ bias) {
  int lane = threadIdx.x & 31, wid = threadIdx.x >> 5;
  int half = lane >> 4, lm = lane & 15;
  // Hoist B = W1^T fragments (uniform across all tiles): B[k][n] = W1[n][k]
  v16h Bf[4][2];
  float b1n[4], w2n[4];
  for (int nt = 0; nt < 4; ++nt) {
    int n = nt * 16 + lm;
    b1n[nt] = b1[n];
    w2n[nt] = w2[n];
    for (int ks = 0; ks < 2; ++ks) {
      v16h bf;
      for (int v = 0; v < 8; ++v) {
        int k = ks * 32 + half * 16 + 2 * v;
        bf[2 * v]     = (_Float16)w1[n * 64 + k];
        bf[2 * v + 1] = (_Float16)w1[n * 64 + k + 1];
      }
      Bf[nt][ks] = bf;
    }
  }
  float b2v = b2[0];
  long wave = (long)blockIdx.x * 4 + wid;      // 10000 waves * 8 tiles = 80000 tiles
#pragma unroll 1
  for (int t = 0; t < CPB_TPW; ++t) {
    long tile = wave * CPB_TPW + t;
    long base = tile * 16;
    long p = base + lm;                        // this lane's point (A-row M = lm)
    int g = (int)(p / (HW * NJ));
    int r1 = (int)(p - (long)g * (HW * NJ));
    int i = r1 / NJ, j = r1 - i * NJ;
    float qx = (float)(i % 40) * (2.0f / 39.0f) - 1.0f;
    float qy = (float)(i / 40) * (2.0f / 39.0f) - 1.0f;
    float px = qx - vgrid[(g * NJ + j) * 2 + 0];
    float py = qy - vgrid[(g * NJ + j) * 2 + 1];
    float bx = copysignf(log1pf(fabsf(px)), px);
    float by = copysignf(log1pf(fabsf(py)), py);
    // A fragments: h0 = relu(w0 @ [bx,by] + b0), built directly in A layout
    v16h Af[2];
    for (int ks = 0; ks < 2; ++ks) {
      v16h af;
      for (int v = 0; v < 8; ++v) {
        int k0 = ks * 32 + akidx(v, half);
        float a0 = fmaxf(0.f, w0[k0 * 2 + 0] * bx + w0[k0 * 2 + 1] * by + b0[k0]);
        float a1 = fmaxf(0.f, w0[(k0 + 1) * 2 + 0] * bx + w0[(k0 + 1) * 2 + 1] * by + b0[k0 + 1]);
        af[2 * v] = (_Float16)a0; af[2 * v + 1] = (_Float16)a1;
      }
      Af[ks] = af;
    }
    float partial[8] = {0, 0, 0, 0, 0, 0, 0, 0};
    for (int nt = 0; nt < 4; ++nt) {
      v8f c = {};
      c = wmma16(Af[0], Bf[nt][0], c);
      c = wmma16(Af[1], Bf[nt][1], c);
      for (int r = 0; r < 8; ++r) partial[r] += fmaxf(0.f, c[r] + b1n[nt]) * w2n[nt];
    }
    for (int m = 1; m < 16; m <<= 1)    // reduce over the 16 n-lanes in each half
      for (int r = 0; r < 8; ++r) partial[r] += __shfl_xor(partial[r], m, 32);
    if (lm == 0) {
      for (int r = 0; r < 8; ++r) bias[base + 8 * half + r] = partial[r] + b2v;
    }
  }
}

// ---------------- K6: fused attention: sim(WMMA) + bias + softmax + attn@v(WMMA) -> out_t f16 [p][c]
__global__ void __launch_bounds__(128, 1)
k_attention(const _Float16* __restrict__ qs, const _Float16* __restrict__ k16,
            const _Float16* __restrict__ vt, const float* __restrict__ bias,
            _Float16* __restrict__ out_t) {
  __shared__ _Float16 att_sh[4][16][JK];
  int h = blockIdx.x / 25;
  int itile = (blockIdx.x % 25) * 64;
  int lane = threadIdx.x & 31, wid = threadIdx.x >> 5;
  int half = lane >> 4, lm = lane & 15;
  int ibase = itile + wid * 16;
  // A (q*scale) fragments
  v16h Aq[2];
  {
    int irow = ibase + lm;
    for (int ks = 0; ks < 2; ++ks) {
      v16h a;
      for (int v = 0; v < 8; ++v) {
        int d = ks * 32 + akidx(v, half);
        const _Float16* s = qs + ((h * HW + irow) * DH + d);
        a[2 * v] = s[0]; a[2 * v + 1] = s[1];
      }
      Aq[ks] = a;
    }
  }
  float s[7][8];
  for (int nt = 0; nt < 7; ++nt) {
    v8f c = {};
    int j = nt * 16 + lm;
    for (int ks = 0; ks < 2; ++ks) {
      v16h b;
      for (int v = 0; v < 8; ++v) {
        int d = ks * 32 + half * 16 + 2 * v;
        const _Float16* src = k16 + ((h * JP + j) * DH + d);
        b[2 * v] = src[0]; b[2 * v + 1] = src[1];
      }
      c = wmma16(Aq[ks], b, c);
    }
    for (int r = 0; r < 8; ++r) {
      int i = ibase + 8 * half + r;
      s[nt][r] = (j < NJ) ? (c[r] + bias[(h * HW + i) * NJ + j]) : -3.0e38f;
    }
  }
  // softmax over j (row i spread across 16 lanes x 7 frags)
  for (int r = 0; r < 8; ++r) {
    float m = s[0][r];
    for (int nt = 1; nt < 7; ++nt) m = fmaxf(m, s[nt][r]);
    for (int msk = 1; msk < 16; msk <<= 1) m = fmaxf(m, __shfl_xor(m, msk, 32));
    float e = 0.f;
    for (int nt = 0; nt < 7; ++nt) { s[nt][r] = __expf(s[nt][r] - m); e += s[nt][r]; }
    for (int msk = 1; msk < 16; msk <<= 1) e += __shfl_xor(e, msk, 32);
    float inv = 1.0f / e;
    for (int nt = 0; nt < 7; ++nt) s[nt][r] *= inv;
  }
  // transpose attn to A layout via LDS (+ zero K-pad columns 112..127)
  for (int nt = 0; nt < 7; ++nt)
    for (int r = 0; r < 8; ++r)
      att_sh[wid][8 * half + r][nt * 16 + lm] = (_Float16)s[nt][r];
  for (int r = 0; r < 8; ++r)
    att_sh[wid][8 * half + r][JP + lm] = (_Float16)0.f;
  __syncthreads();
  // out = attn @ v  (K = 128 padded)
  for (int dt = 0; dt < 4; ++dt) {
    v8f c = {};
    int d = dt * 16 + lm;
    for (int ks = 0; ks < 4; ++ks) {
      v16h a, b;
      for (int v = 0; v < 8; ++v) {
        int ja = ks * 32 + akidx(v, half);
        a[2 * v] = att_sh[wid][lm][ja];
        a[2 * v + 1] = att_sh[wid][lm][ja + 1];
        int jb = ks * 32 + half * 16 + 2 * v;
        const _Float16* src = vt + ((h * DH + d) * JK + jb);
        b[2 * v] = src[0]; b[2 * v + 1] = src[1];
      }
      c = wmma16(a, b, c);
    }
    for (int r = 0; r < 8; ++r) {
      int i = ibase + 8 * half + r;
      out_t[i * 512 + h * DH + d] = (_Float16)c[r];
    }
  }
}

// ---------------- K7: f32 -> f16 convert (w_out)
__global__ void k_cvt16(const float* __restrict__ w, _Float16* __restrict__ w16, int n) {
  int idx = blockIdx.x * 256 + threadIdx.x;
  if (idx < n) w16[idx] = (_Float16)w[idx];
}

// ---------------- K8: final projection D[o][p] = sum_c w_out[o][c]*out_t[p][c] + b_out[o]
__global__ void __launch_bounds__(128, 1)
k_final_proj(const _Float16* __restrict__ wout16, const _Float16* __restrict__ out_t,
             const float* __restrict__ b_out, float* __restrict__ out) {
  int lane = threadIdx.x & 31, wid = threadIdx.x >> 5;
  int half = lane >> 4, lm = lane & 15;
  int tile = blockIdx.x * 4 + wid;   // 16 M-tiles x 100 N-tiles = 1600
  int mt = tile / 100, ntp = tile % 100;
  int o = mt * 16 + lm, p = ntp * 16 + lm;
  v8f c = {};
  for (int ks = 0; ks < 16; ++ks) {
    v16h a, b;
    for (int v = 0; v < 8; ++v) {
      int ca = ks * 32 + akidx(v, half);
      const _Float16* sa = wout16 + (o * 512 + ca);
      a[2 * v] = sa[0]; a[2 * v + 1] = sa[1];
      int cb = ks * 32 + half * 16 + 2 * v;
      const _Float16* sb = out_t + (p * 512 + cb);
      b[2 * v] = sb[0]; b[2 * v + 1] = sb[1];
    }
    c = wmma16(a, b, c);
  }
  for (int r = 0; r < 8; ++r) {
    int oo = mt * 16 + 8 * half + r;
    int pp = ntp * 16 + lm;
    out[oo * HW + pp] = c[r] + b_out[oo];
  }
}

extern "C" void kernel_launch(void* const* d_in, const int* in_sizes, int n_in,
                              void* d_out, int out_size, void* d_ws, size_t ws_size,
                              hipStream_t stream) {
  (void)in_sizes; (void)n_in; (void)out_size; (void)ws_size;
  const int L = 5;  // only the last layer's output survives the encoder loop
  const float* x      = (const float*)d_in[0];
  const float* wq     = (const float*)d_in[1]  + (size_t)L * 512 * 32;
  const float* wk     = (const float*)d_in[2]  + (size_t)L * 512 * 32;
  const float* wv     = (const float*)d_in[3]  + (size_t)L * 512 * 32;
  const float* w_off1 = (const float*)d_in[4]  + (size_t)L * 64 * 36;
  const float* b_off1 = (const float*)d_in[5]  + (size_t)L * 64;
  const float* w_off2 = (const float*)d_in[6]  + (size_t)L * 2 * 64;
  const float* cpb_w0 = (const float*)d_in[7]  + (size_t)L * 64 * 2;
  const float* cpb_b0 = (const float*)d_in[8]  + (size_t)L * 64;
  const float* cpb_w1 = (const float*)d_in[9]  + (size_t)L * 64 * 64;
  const float* cpb_b1 = (const float*)d_in[10] + (size_t)L * 64;
  const float* cpb_w2 = (const float*)d_in[11] + (size_t)L * 64;
  const float* cpb_b2 = (const float*)d_in[12] + (size_t)L * 1;
  const float* w_out  = (const float*)d_in[13] + (size_t)L * 256 * 512;
  const float* b_out  = (const float*)d_in[14] + (size_t)L * 256;

  char* ws = (char*)d_ws;
  size_t off = 0;
  auto carve = [&](size_t bytes) { char* p = ws + off; off = (off + bytes + 255) & ~(size_t)255; return p; };
  float*    q_f32 = (float*)   carve((size_t)512 * HW * 4);
  _Float16* qs16  = (_Float16*)carve((size_t)NH * HW * DH * 2);
  float*    vgrid = (float*)   carve((size_t)NH * NJ * 2 * 4);
  float*    kv    = (float*)   carve((size_t)256 * NJ * 4);
  _Float16* k16   = (_Float16*)carve((size_t)NH * JP * DH * 2);
  _Float16* vt16  = (_Float16*)carve((size_t)NH * DH * JK * 2);
  float*    bias  = (float*)   carve((size_t)NH * HW * NJ * 4);
  _Float16* out_t = (_Float16*)carve((size_t)HW * 512 * 2);
  _Float16* wo16  = (_Float16*)carve((size_t)256 * 512 * 2);

  k_proj_q   <<<3200, 256, 0, stream>>>(x, wq, q_f32, qs16);
  k_offsets  <<<800,   64, 0, stream>>>(q_f32, w_off1, b_off1, w_off2, vgrid);
  k_sample   <<<100,  256, 0, stream>>>(x, vgrid, kv);
  k_proj_kv  <<<480,  256, 0, stream>>>(kv, wk, wv, k16, vt16);
  k_cpb_bias <<<2500, 128, 0, stream>>>(vgrid, cpb_w0, cpb_b0, cpb_w1, cpb_b1,
                                        cpb_w2, cpb_b2, bias);
  k_cvt16    <<<512,  256, 0, stream>>>(w_out, wo16, 256 * 512);
  k_attention<<<200,  128, 0, stream>>>(qs16, k16, vt16, bias, out_t);
  k_final_proj<<<400, 128, 0, stream>>>(wo16, out_t, b_out, (float*)d_out);
}
